// Net_35416300323255
// MI455X (gfx1250) — matrix operand order repaired and checked by reading, not compiled, not run on hardware
//
#include <hip/hip_runtime.h>
#include <hip/hip_bf16.h>

// ---------------------------------------------------------------------------
// Problem constants (match reference)
// ---------------------------------------------------------------------------
#define INPUT_N   63
#define EMB       3
#define FEAT      (EMB * INPUT_N)   // 189
#define HID       256
#define GATES     (4 * HID)         // 1024
#define SEQ       2048
#define STEPS     INPUT_N           // 63
#define OUTC      24
#define FUN       8
#define EPSV      1e-5f

#define NTHREADS  512
#define NWAVES    16                // 512 / 32

typedef __attribute__((ext_vector_type(16))) _Float16 v16h;
typedef __attribute__((ext_vector_type(8)))  _Float16 v8h;
typedef __attribute__((ext_vector_type(8)))  float    v8f;

// ---------------------------------------------------------------------------
// Helpers
// ---------------------------------------------------------------------------
__device__ __forceinline__ float sigm(float x) { return 1.0f / (1.0f + __expf(-x)); }

__device__ __forceinline__ float rng_uniform(unsigned s) {
  s ^= s >> 16; s *= 0x7feb352du;
  s ^= s >> 15; s *= 0x846ca68bu;
  s ^= s >> 16;
  return (float)(s >> 8) * (1.0f / 16777216.0f) + 1e-12f;
}
__device__ __forceinline__ float gumbel(unsigned seed, int i, int t, int c) {
  unsigned key = seed * 0x9E3779B1u ^ (unsigned)i * 0x85ebca77u
               ^ (unsigned)t * 0xC2B2AE3Du ^ (unsigned)c * 0x27d4eb2fu;
  float u = rng_uniform(key);
  return -__logf(-__logf(u));
}

// Load B-operand tile for v_wmma_f32_16x16x32_f16.
// B is 32x16 (KxN); column n = weight row j.  Lane n (0..15): K=0..7 and
// K=16..23 of column n; lane n+16: K=8..15 and K=24..31.  Row-major f16
// weights -> two 16-byte chunks per lane at W[j*256 + k0 + hi*8 (+16)].
__device__ __forceinline__ v16h load_b_tile(const _Float16* __restrict__ W,
                                            int j, int k0, int hi) {
  const v8h* p1 = (const v8h*)(W + (size_t)j * HID + k0 + hi * 8);
  const v8h* p2 = (const v8h*)(W + (size_t)j * HID + k0 + 16 + hi * 8);
  v8h lo = p1[0], hh = p2[0];
  return __builtin_shufflevector(lo, hh, 0,1,2,3,4,5,6,7,8,9,10,11,12,13,14,15);
}

// Load A-operand with all 16 rows replicated to the vector h (matvec mode).
// 16-bit A 16x32 layout: lane (hi=0): K=0..7 in v0-3, K=16..23 in v4-7;
// lane (hi=1): K=8..15 and K=24..31.
__device__ __forceinline__ v16h load_a_rep(const _Float16* __restrict__ h,
                                           int k0, int hi) {
  const v8h* p1 = (const v8h*)(h + k0 + hi * 8);
  const v8h* p2 = (const v8h*)(h + k0 + 16 + hi * 8);
  v8h lo = p1[0], hh = p2[0];
  return __builtin_shufflevector(lo, hh, 0,1,2,3,4,5,6,7,8,9,10,11,12,13,14,15);
}

// y[j] accumulation for one 16-output tile: acc += (h as 16x32 rows) x W^T
__device__ __forceinline__ v8f wmma_matvec(const _Float16* __restrict__ W,
                                           const _Float16* __restrict__ h,
                                           int j, int hi, v8f acc) {
#pragma unroll
  for (int k0 = 0; k0 < HID; k0 += 32) {
    v16h a = load_a_rep(h, k0, hi);
    v16h b = load_b_tile(W, j, k0, hi);
    acc = __builtin_amdgcn_wmma_f32_16x16x32_f16(false, a, false, b,
                                                 (short)0, acc, false, false);
  }
  return acc;
}

// ---------------------------------------------------------------------------
// Prep kernel 1: f32 -> f16 weight conversion, W_fc zero-pad to 32 rows,
// bias folding.
// ---------------------------------------------------------------------------
__global__ void prep_convert(const float* __restrict__ Whh0,
                             const float* __restrict__ Wih1,
                             const float* __restrict__ Whh1,
                             const float* __restrict__ Wfc,
                             const float* __restrict__ bih0,
                             const float* __restrict__ bhh0,
                             const float* __restrict__ bih1,
                             const float* __restrict__ bhh1,
                             _Float16* __restrict__ Whh0h,
                             _Float16* __restrict__ Wih1h,
                             _Float16* __restrict__ Whh1h,
                             _Float16* __restrict__ Wfch,
                             float* __restrict__ bsum0,
                             float* __restrict__ bsum1) {
  int idx = blockIdx.x * blockDim.x + threadIdx.x;
  int stride = gridDim.x * blockDim.x;
  const int n = GATES * HID;
  for (int p = idx; p < n; p += stride) {
    Whh0h[p] = (_Float16)Whh0[p];
    Wih1h[p] = (_Float16)Wih1[p];
    Whh1h[p] = (_Float16)Whh1[p];
  }
  for (int p = idx; p < 32 * HID; p += stride) {
    int j = p >> 8, k = p & 255;
    Wfch[p] = (j < OUTC) ? (_Float16)Wfc[j * HID + k] : (_Float16)0.0f;
  }
  for (int p = idx; p < GATES; p += stride) {
    bsum0[p] = bih0[p] + bhh0[p];
    bsum1[p] = bih1[p] + bhh1[p];
  }
}

// ---------------------------------------------------------------------------
// Prep kernel 2: per outer step i, the LSTM input is ONE broadcast row.
// G0_all[i][j] = bsum0[j] + sum_k Xv[i][k] * W_ih0[j][k]
// Xv[i][k]     = b_in[k] + sum_m (embed[PAD][m%3] + pos[i][m]) * W_in[k][m]
// ---------------------------------------------------------------------------
__global__ void prep_g0(const float* __restrict__ embed,
                        const float* __restrict__ pos,
                        const float* __restrict__ Win,
                        const float* __restrict__ bin,
                        const float* __restrict__ Wih0,
                        const float* __restrict__ bsum0,
                        float* __restrict__ G0all) {
  __shared__ float xin[FEAT + 4];
  __shared__ float xv[INPUT_N + 1];
  const int i = blockIdx.x;
  const int tid = threadIdx.x;
  if (tid < FEAT) xin[tid] = embed[tid % EMB] + pos[(size_t)i * FEAT + tid];
  __syncthreads();
  if (tid < INPUT_N) {
    float s = bin[tid];
    const float* wr = Win + (size_t)tid * FEAT;
    for (int m = 0; m < FEAT; ++m) s += xin[m] * wr[m];
    xv[tid] = s;
  }
  __syncthreads();
  for (int j = tid; j < GATES; j += blockDim.x) {
    float s = bsum0[j];
    const float* wr = Wih0 + (size_t)j * INPUT_N;
    for (int k = 0; k < INPUT_N; ++k) s += xv[k] * wr[k];
    G0all[(size_t)i * GATES + j] = s;
  }
}

// ---------------------------------------------------------------------------
// Main persistent kernel: one workgroup owns the whole recurrence.
// ---------------------------------------------------------------------------
__global__ __launch_bounds__(NTHREADS)
void lstm_seq_kernel(const float* __restrict__ G0all,
                     const float* __restrict__ bsum1,
                     const _Float16* __restrict__ Whh0h,
                     const _Float16* __restrict__ Wih1h,
                     const _Float16* __restrict__ Whh1h,
                     const _Float16* __restrict__ Wfch,
                     const float* __restrict__ bfc,
                     const float* __restrict__ gamma_g,
                     const float* __restrict__ beta_g,
                     const float* __restrict__ h0in,
                     const float* __restrict__ c0in,
                     float* __restrict__ h1out,
                     float* __restrict__ dout,
                     const int* __restrict__ seedp) {
  __shared__ float sh_h0[HID], sh_c0[HID], sh_h1[HID], sh_c1[HID];
  __shared__ __align__(32) _Float16 sh_h0h[HID];
  __shared__ __align__(32) _Float16 sh_h1h[HID];
  __shared__ float sh_g[GATES];
  __shared__ float sh_g0[GATES];     // per-outer-step input gates (LDS cached)
  __shared__ float sh_bs1[GATES];    // layer-1 bias sum (LDS cached)
  __shared__ float sh_bnsum[HID], sh_bnsq[HID];
  __shared__ float sh_sA[HID], sh_sB[HID];
  __shared__ float sh_gam[HID], sh_bet[HID];
  __shared__ float sh_bfc[OUTC];
  __shared__ float sh_log[NWAVES][16][32];

  const int tid  = threadIdx.x;
  const int lane = tid & 31;
  const int wv   = tid >> 5;
  const int hi   = lane >> 4;
  const int ln   = lane & 15;
  const unsigned seed = (unsigned)seedp[0];

  if (tid < HID) {
    sh_h0[tid] = h0in[tid];        sh_c0[tid] = c0in[tid];
    sh_h1[tid] = h0in[HID + tid];  sh_c1[tid] = c0in[HID + tid];
    sh_h0h[tid] = (_Float16)sh_h0[tid];
    sh_h1h[tid] = (_Float16)sh_h1[tid];
    sh_bnsum[tid] = 0.0f; sh_bnsq[tid] = 0.0f;
    sh_gam[tid] = gamma_g[tid]; sh_bet[tid] = beta_g[tid];
  }
  if (tid < OUTC) sh_bfc[tid] = bfc[tid];
  for (int j = tid; j < GATES; j += NTHREADS) sh_bs1[j] = bsum1[j];
  __syncthreads();

  for (int i = 0; i < STEPS; ++i) {
    // Stage this outer step's precomputed input gates into LDS (reused 2048x).
    for (int j = tid; j < GATES; j += NTHREADS)
      sh_g0[j] = G0all[(size_t)i * GATES + j];
    __syncthreads();

    // -------- Phase 1: sequential LSTM over SEQ steps --------
    for (int t = 0; t < SEQ; ++t) {
      // layer 0 gates: g = W_hh0 @ h0 + (precomputed input + biases)
      for (int tile = wv; tile < GATES / 16; tile += NWAVES) {
        int j = tile * 16 + ln;
        v8f acc = {};
        acc = wmma_matvec(Whh0h, sh_h0h, j, hi, acc);
        if (hi == 0) sh_g[j] = acc[0] + sh_g0[j];
      }
      __syncthreads();
      if (tid < HID) {
        float gi = sh_g[tid],          gf = sh_g[HID + tid];
        float gg = sh_g[2 * HID + tid], go = sh_g[3 * HID + tid];
        float c = sigm(gf) * sh_c0[tid] + sigm(gi) * tanhf(gg);
        float h = sigm(go) * tanhf(c);
        sh_c0[tid] = c; sh_h0[tid] = h; sh_h0h[tid] = (_Float16)h;
      }
      __syncthreads();
      // layer 1 gates: g = W_ih1 @ h0' + W_hh1 @ h1 + bsum1
      for (int tile = wv; tile < GATES / 16; tile += NWAVES) {
        int j = tile * 16 + ln;
        v8f acc = {};
        acc = wmma_matvec(Wih1h, sh_h0h, j, hi, acc);
        acc = wmma_matvec(Whh1h, sh_h1h, j, hi, acc);
        if (hi == 0) sh_g[j] = acc[0] + sh_bs1[j];
      }
      __syncthreads();
      if (tid < HID) {
        float gi = sh_g[tid],          gf = sh_g[HID + tid];
        float gg = sh_g[2 * HID + tid], go = sh_g[3 * HID + tid];
        float c = sigm(gf) * sh_c1[tid] + sigm(gi) * tanhf(gg);
        float h = sigm(go) * tanhf(c);
        sh_c1[tid] = c; sh_h1[tid] = h; sh_h1h[tid] = (_Float16)h;
        h1out[(size_t)t * HID + tid] = h;
        sh_bnsum[tid] += h;
        sh_bnsq[tid]  += h * h;
      }
      __syncthreads();
    }

    // -------- BN stats (training mode, biased) --------
    __threadfence();
    if (tid < HID) {
      float mu  = sh_bnsum[tid] * (1.0f / SEQ);
      float var = sh_bnsq[tid] * (1.0f / SEQ) - mu * mu;
      float rs  = rsqrtf(var + EPSV);
      float A   = sh_gam[tid] * rs;
      sh_sA[tid] = A;
      sh_sB[tid] = sh_bet[tid] - A * mu;
      sh_bnsum[tid] = 0.0f; sh_bnsq[tid] = 0.0f;
      // warm L2/L0 for the next outer step's input-gate row (cold path)
      if (i + 1 < STEPS)
        __builtin_prefetch(G0all + (size_t)(i + 1) * GATES + tid, 0, 1);
    }
    __syncthreads();

    // -------- Phase 2: logits GEMM (2048x24, K=256) + sampling --------
    for (int mt = wv; mt < SEQ / 16; mt += NWAVES) {
      int row = mt * 16 + ln;                 // timestep for this lane's row
      const float* hrow = h1out + (size_t)row * HID;
#pragma unroll
      for (int nt = 0; nt < 2; ++nt) {
        v8f acc = {};
#pragma unroll
        for (int kk = 0; kk < 8; ++kk) {
          int kb = kk * 32;
          v16h a;
#pragma unroll
          for (int e = 0; e < 8; ++e) {
            int k1 = kb + hi * 8 + e;
            int k2 = kb + 16 + hi * 8 + e;
            float y1 = sh_sA[k1] * hrow[k1] + sh_sB[k1];
            float y2 = sh_sA[k2] * hrow[k2] + sh_sB[k2];
            a[e]     = (_Float16)__expf(-y1 * y1);
            a[8 + e] = (_Float16)__expf(-y2 * y2);
          }
          v16h b = load_b_tile(Wfch, nt * 16 + ln, kb, hi);
          acc = __builtin_amdgcn_wmma_f32_16x16x32_f16(false, a, false, b,
                                                       (short)0, acc, false, false);
        }
        int col = nt * 16 + ln;
#pragma unroll
        for (int r = 0; r < 8; ++r) sh_log[wv][hi * 8 + r][col] = acc[r];
      }
      __syncthreads();   // uniform: every wave runs the same trip counts
      if (hi == 0) {
        int t = row;
        float lg[OUTC];
        float mx = -1e30f;
        for (int c = 0; c < OUTC; ++c) {
          float v = sh_log[wv][ln][c] + sh_bfc[c];
          lg[c] = v; mx = fmaxf(mx, v);
        }
        float s = 0.0f;
        for (int c = 0; c < OUTC; ++c) s += __expf(lg[c] - mx);
        float lse = mx + __logf(s);
        // Gumbel-max categorical (full head / tail-only)
        int afull = 0; float bf = -1e30f;
        for (int c = 0; c < OUTC; ++c) {
          float v = lg[c] + gumbel(seed, i, t, c);
          if (v > bf) { bf = v; afull = c; }
        }
        int atail = FUN; float bt = -1e30f;
        for (int c = FUN; c < OUTC; ++c) {
          float v = lg[c] + gumbel(seed, i, t, c + 64);
          if (v > bt) { bt = v; atail = c; }
        }
        int act = (i < (INPUT_N - 1) / 2) ? afull : atail;   // head_len = 31
        dout[(size_t)i * SEQ + t] = (float)act;
        dout[(size_t)STEPS * SEQ + (size_t)i * SEQ + t] = lg[act] - lse;
      }
      __syncthreads();
    }
    __syncthreads();
  }
}

// ---------------------------------------------------------------------------
// Launch
// ---------------------------------------------------------------------------
extern "C" void kernel_launch(void* const* d_in, const int* in_sizes, int n_in,
                              void* d_out, int out_size, void* d_ws, size_t ws_size,
                              hipStream_t stream) {
  (void)in_sizes; (void)n_in; (void)out_size; (void)ws_size;

  const float* embed = (const float*)d_in[0];
  const float* pos   = (const float*)d_in[1];
  const float* Win   = (const float*)d_in[2];
  const float* bin   = (const float*)d_in[3];
  const float* Wih0  = (const float*)d_in[4];
  const float* Whh0  = (const float*)d_in[5];
  const float* bih0  = (const float*)d_in[6];
  const float* bhh0  = (const float*)d_in[7];
  const float* Wih1  = (const float*)d_in[8];
  const float* Whh1  = (const float*)d_in[9];
  const float* bih1  = (const float*)d_in[10];
  const float* bhh1  = (const float*)d_in[11];
  const float* gamma = (const float*)d_in[12];
  const float* beta  = (const float*)d_in[13];
  const float* Wfc   = (const float*)d_in[14];
  const float* bfc   = (const float*)d_in[15];
  const float* h0    = (const float*)d_in[16];
  const float* c0    = (const float*)d_in[17];
  const int*   seedp = (const int*)d_in[18];

  char* ws = (char*)d_ws;
  const size_t WBYTES = (size_t)GATES * HID * sizeof(_Float16);  // 512 KB each
  _Float16* Whh0h = (_Float16*)(ws + 0);
  _Float16* Wih1h = (_Float16*)(ws + WBYTES);
  _Float16* Whh1h = (_Float16*)(ws + 2 * WBYTES);
  _Float16* Wfch  = (_Float16*)(ws + 3 * WBYTES);                // 16 KB
  char* p = ws + 3 * WBYTES + 32 * HID * sizeof(_Float16);
  float* bsum0 = (float*)p;                 p += GATES * sizeof(float);
  float* bsum1 = (float*)p;                 p += GATES * sizeof(float);
  float* G0all = (float*)p;                 p += (size_t)STEPS * GATES * sizeof(float);
  float* h1out = (float*)p;                 // SEQ*HID floats = 2 MB

  prep_convert<<<256, 256, 0, stream>>>(Whh0, Wih1, Whh1, Wfc,
                                        bih0, bhh0, bih1, bhh1,
                                        Whh0h, Wih1h, Whh1h, Wfch,
                                        bsum0, bsum1);
  prep_g0<<<STEPS, 256, 0, stream>>>(embed, pos, Win, bin, Wih0, bsum0, G0all);
  lstm_seq_kernel<<<1, NTHREADS, 0, stream>>>(G0all, bsum1,
                                              Whh0h, Wih1h, Whh1h, Wfch,
                                              bfc, gamma, beta, h0, c0,
                                              h1out, (float*)d_out, seedp);
}
